// StateSpaceSimulatorAugmented_Non_normalized_120259084395
// MI455X (gfx1250) — matrix operand order repaired and checked by reading, not compiled, not run on hardware
//
#include <hip/hip_runtime.h>
#include <hip/hip_bf16.h>

// MI455X / gfx1250, wave32. One wave (32 threads) per block handles 16 batch
// rows for all 512 timesteps. All matrix math via v_wmma_f32_16x16x32_f16.

typedef __attribute__((ext_vector_type(16))) _Float16 v16h;
typedef __attribute__((ext_vector_type(8)))  _Float16 v8h;
typedef __attribute__((ext_vector_type(8)))  float    v8f;

namespace {
constexpr int kB = 2048, kT = 512;
constexpr int FRAG_H = 512;            // halves per B-fragment (32 lanes * 16)
// fragment table offsets (units of fragments)
constexpr int F_W1A  = 0;   // 24: phase-A layer1, 64K x 192N (f|f_b|g_b), idx = kt*12+nt
constexpr int F_W2AF = 24;  // 4 : layer2 f+f_b fused, 128K x 16N
constexpr int F_WGB  = 28;  // 2 : layer2 g_b, 64K x 16N
constexpr int F_W1B  = 30;  // 16: phase-B layer1, 64K x 128N (h|h_b), idx = kt*8+nt
constexpr int F_W2B  = 46;  // 4 : layer2 h+h_b fused, 128K x 16N (out padded 8->16)
constexpr int N_FRAG = 50;

struct SimArgs {
  const float *x_f_0, *x_b_0, *u, *theta_f, *theta_h;
  const float *min_xb, *max_xb, *min_x, *max_x, *min_y, *max_y;
  const float *u_mean, *u_std, *x_mean, *x_std, *y_mean, *y_std;
  const float *fW1,  *fb1,  *fW2,  *fb2;
  const float *fbW1, *fbb1, *fbW2, *fbb2;
  const float *gbW1, *gbb1, *gbW2, *gbb2;
  const float *hW1,  *hb1,  *hW2,  *hb2;
  const float *hbW1, *hbb1, *hbW2, *hbb2;
  float *oXF, *oXB, *oY;
};
} // namespace

// ---- weight element gather: (k,n) of the fused virtual matrices -------------
// unified phase-A input: [xf 0-15 | xb 16-31 | u 32-39 | theta_f 40-43 | 0.. | 1.0 @63]
__device__ __forceinline__ float w1a_elem(int k, int n, const SimArgs& A) {
  int mlp = n >> 6, c = n & 63;
  if (mlp == 0) {                       // f: input [xf(16), u(8), theta_f(4)]
    if (k < 16)            return A.fW1[k * 64 + c];
    if (k >= 32 && k < 44) return A.fW1[(k - 16) * 64 + c]; // u rows 16-23, tf 24-27
    if (k == 63)           return A.fb1[c];
    return 0.f;
  }
  const float* W = (mlp == 1) ? A.fbW1 : A.gbW1;  // f_b / g_b: input [xf,xb,u](40)
  const float* b = (mlp == 1) ? A.fbb1 : A.gbb1;
  if (k < 40)  return W[k * 64 + c];
  if (k == 63) return b[c];
  return 0.f;
}
__device__ __forceinline__ float w2af_elem(int k, int n, const SimArgs& A) {
  return (k < 64) ? A.fW2[k * 16 + n] : A.fbW2[(k - 64) * 16 + n];
}
// unified phase-B input: [xf2 0-15 | xb2 16-31 | theta_h 32-35 | 0.. | 1.0 @63]
__device__ __forceinline__ float w1b_elem(int k, int n, const SimArgs& A) {
  int mlp = n >> 6, c = n & 63;
  if (mlp == 0) {                       // h: input [xf(16), theta_h(4)]
    if (k < 16)            return A.hW1[k * 64 + c];
    if (k >= 32 && k < 36) return A.hW1[(k - 16) * 64 + c];
    if (k == 63)           return A.hb1[c];
    return 0.f;
  }
  if (k < 32)  return A.hbW1[k * 64 + c];           // h_b: input [xf,xb](32)
  if (k == 63) return A.hbb1[c];
  return 0.f;
}
__device__ __forceinline__ float w2b_elem(int k, int n, const SimArgs& A) {
  if (n >= 8) return 0.f;                           // pad output 8 -> 16
  return (k < 64) ? A.hW2[k * 8 + n] : A.hbW2[(k - 64) * 8 + n];
}

// ---- WMMA helpers -----------------------------------------------------------
__device__ __forceinline__ v8f wmma16(v16h a, v16h b, v8f c) {
  return __builtin_amdgcn_wmma_f32_16x16x32_f16(
      /*neg_a=*/false, a, /*neg_b=*/false, b, /*c_mod=*/(short)0, c,
      /*reuse_a=*/false, /*reuse_b=*/false);
}
// B fragment: pre-packed in LDS, contiguous 32B per lane -> 2x ds_load_b128
__device__ __forceinline__ v16h ld_bfrag(const _Float16* sFrag, int frag, int lane) {
  return *(const v16h*)(sFrag + frag * FRAG_H + lane * 16);
}
// A fragment from a row-major f16 staging buffer (ld halves per row).
// ISA layout: lanes 0-15 hold row m, K = k0+{0..7,16..23}; lanes 16-31 get +8.
__device__ __forceinline__ v16h ld_afrag(const _Float16* buf, int k0, int ld, int lane) {
  int m  = lane & 15;
  int kb = k0 + ((lane >> 4) << 3);
  const v8h* p = (const v8h*)(buf + m * ld + kb);
  v8h lo = p[0];
  v8h hh = p[2];      // +16 halves
  v16h a;
#pragma unroll
  for (int i = 0; i < 8; ++i) { a[i] = lo[i]; a[i + 8] = hh[i]; }
  return a;
}
__device__ __forceinline__ float fast_tanh(float x) {
  float e2 = __expf(2.0f * x);                       // v_exp path
  return (e2 - 1.0f) * __builtin_amdgcn_rcpf(e2 + 1.0f);
}

__global__ __launch_bounds__(32)
void sim_ssm_wmma_kernel(SimArgs A) {
  __shared__ __align__(16) _Float16 sFrag[N_FRAG * FRAG_H]; // 51200 B weights
  __shared__ __align__(16) _Float16 sInA[16 * 64];          // phase-A input staging
  __shared__ __align__(16) _Float16 sInB[16 * 64];          // phase-B input staging
  __shared__ __align__(16) _Float16 sHid[16 * 192];         // hidden staging (shared A/B)

  const int lane = threadIdx.x & 31;
  const int tile = blockIdx.x;              // 16 batch rows per tile
  const int n    = lane & 15;               // feature column in C/D layout
  const int hi   = lane >> 4;               // row-half selector

  // ---- one-time: pack all weights (fp32 global) into f16 B-fragments in LDS
  for (int fr = 0; fr < N_FRAG; ++fr) {
#pragma unroll
    for (int h = 0; h < 16; ++h) {
      int kin = (h & 7) + (hi << 3) + ((h >> 3) << 4); // B layout: lanes 0-15 K={0..7,16..23}, 16-31 get +8
      float v;
      if (fr < F_W2AF)     { int kt = fr / 12, nt = fr % 12;
                             v = w1a_elem(kt * 32 + kin, nt * 16 + n, A); }
      else if (fr < F_WGB) { int kt = fr - F_W2AF; v = w2af_elem(kt * 32 + kin, n, A); }
      else if (fr < F_W1B) { int kt = fr - F_WGB;  v = A.gbW2[(kt * 32 + kin) * 16 + n]; }
      else if (fr < F_W2B) { int i = fr - F_W1B;
                             v = w1b_elem((i >> 3) * 32 + kin, (i & 7) * 16 + n, A); }
      else                 { int kt = fr - F_W2B; v = w2b_elem(kt * 32 + kin, n, A); }
      sFrag[fr * FRAG_H + lane * 16 + h] = (_Float16)v;
    }
  }

  // ---- one-time: constant columns of the staging buffers (theta, bias-1)
  if (lane < 16) {
    int b = tile * 16 + lane;
    for (int c = 32; c < 64; ++c) {
      sInA[lane * 64 + c] = (_Float16)0.f;
      sInB[lane * 64 + c] = (_Float16)0.f;
    }
#pragma unroll
    for (int c = 0; c < 4; ++c) {
      sInA[lane * 64 + 40 + c] = (_Float16)A.theta_f[b * 4 + c];
      sInB[lane * 64 + 32 + c] = (_Float16)A.theta_h[b * 4 + c];
    }
    sInA[lane * 64 + 63] = (_Float16)1.f;   // bias-fold input
    sInB[lane * 64 + 63] = (_Float16)1.f;
  }

  // ---- per-lane feature constants (column n of C/D layout)
  const float xmean = A.x_mean[n], xstd = A.x_std[n];
  const float rxstd = 1.0f / xstd;
  const float minx  = A.min_x[n] * xstd + xmean;
  const float maxx  = A.max_x[n] * xstd + xmean;
  const float minxb = A.min_xb[n], maxxb = A.max_xb[n];
  const int   nyc   = n & 7;                 // y columns (only n<8 valid/stored)
  const float ymean = A.y_mean[nyc], ystd = A.y_std[nyc];
  const float rystd = 1.0f / ystd;
  const float miny  = A.min_y[nyc] * ystd + ymean;
  const float maxy  = A.max_y[nyc] * ystd + ymean;
  const float b2ff  = A.fb2[n] + A.fbb2[n];  // fused layer-2 biases
  const float b2gb  = A.gbb2[n];
  const float b2y   = (n < 8) ? (A.hb2[n] + A.hbb2[n]) : 0.f;

  float um[8], us[8];
#pragma unroll
  for (int c = 0; c < 8; ++c) { um[c] = A.u_mean[c]; us[c] = A.u_std[c]; }

  // ---- state in f32 accumulator layout: lane holds (m = hi*8+v, n)
  v8f xf, xb, y;
#pragma unroll
  for (int v = 0; v < 8; ++v) {
    int b = tile * 16 + hi * 8 + v;
    xf[v] = A.x_f_0[b * 16 + n] * xstd + xmean;    // denorm initial xf
    xb[v] = A.x_b_0[b * 16 + n];
  }

  // phase B: y = h(xf2, theta_h) + h_b(xf2, xb2), both layers via WMMA
  auto phaseB = [&](v8f xf2, v8f xb2) -> v8f {
#pragma unroll
    for (int v = 0; v < 8; ++v) {
      int m = hi * 8 + v;
      sInB[m * 64 + n]      = (_Float16)xf2[v];
      sInB[m * 64 + 16 + n] = (_Float16)xb2[v];
    }
    v16h a0 = ld_afrag(sInB, 0, 64, lane);
    v16h a1 = ld_afrag(sInB, 32, 64, lane);
#pragma unroll
    for (int nt = 0; nt < 8; ++nt) {
      v8f acc = {};
      acc = wmma16(a0, ld_bfrag(sFrag, F_W1B + nt, lane), acc);
      acc = wmma16(a1, ld_bfrag(sFrag, F_W1B + 8 + nt, lane), acc);
#pragma unroll
      for (int v = 0; v < 8; ++v) {
        sHid[(hi * 8 + v) * 192 + nt * 16 + n] = (_Float16)fast_tanh(acc[v]);
      }
    }
    v8f cy;
#pragma unroll
    for (int v = 0; v < 8; ++v) cy[v] = b2y;
#pragma unroll
    for (int kt = 0; kt < 4; ++kt)
      cy = wmma16(ld_afrag(sHid, kt * 32, 192, lane),
                  ld_bfrag(sFrag, F_W2B + kt, lane), cy);
    return cy;
  };

  y = phaseB(xf, xb);                        // y0 (unclipped, matches reference)

#pragma unroll 1
  for (int t = 0; t < kT; ++t) {
    // 1) record carry (state BEFORE update)
#pragma unroll
    for (int v = 0; v < 8; ++v) {
      long b = tile * 16 + hi * 8 + v;
      A.oXF[(b * kT + t) * 16 + n] = (xf[v] - xmean) * rxstd;
      A.oXB[(b * kT + t) * 16 + n] = xb[v];
    }
    if (n < 8) {
#pragma unroll
      for (int v = 0; v < 8; ++v) {
        long b = tile * 16 + hi * 8 + v;
        A.oY[(b * kT + t) * 8 + n] = (y[v] - ymean) * rystd;
      }
    }
    // 2) stage phase-A inputs: state + denormalized u_t
#pragma unroll
    for (int v = 0; v < 8; ++v) {
      int m = hi * 8 + v;
      sInA[m * 64 + n]      = (_Float16)xf[v];
      sInA[m * 64 + 16 + n] = (_Float16)xb[v];
    }
    if (lane < 16) {
      const float* up = A.u + ((long)(tile * 16 + lane) * kT + t) * 8;
#pragma unroll
      for (int c = 0; c < 8; ++c)
        sInA[lane * 64 + 32 + c] = (_Float16)(up[c] * us[c] + um[c]);
    }
    // 3) fused layer-1 of f / f_b / g_b : [16x64] x [64x192] + tanh
    v16h a0 = ld_afrag(sInA, 0, 64, lane);
    v16h a1 = ld_afrag(sInA, 32, 64, lane);
#pragma unroll
    for (int nt = 0; nt < 12; ++nt) {
      v8f acc = {};
      acc = wmma16(a0, ld_bfrag(sFrag, F_W1A + nt, lane), acc);
      acc = wmma16(a1, ld_bfrag(sFrag, F_W1A + 12 + nt, lane), acc);
#pragma unroll
      for (int v = 0; v < 8; ++v)
        sHid[(hi * 8 + v) * 192 + nt * 16 + n] = (_Float16)fast_tanh(acc[v]);
    }
    // 4) layer-2: xf2 = xf + b2 + [hid_f|hid_fb] x [W2_f;W2_fb]  (C = state!)
    v8f xf2, xb2;
#pragma unroll
    for (int v = 0; v < 8; ++v) { xf2[v] = xf[v] + b2ff; xb2[v] = xb[v] + b2gb; }
#pragma unroll
    for (int kt = 0; kt < 4; ++kt)
      xf2 = wmma16(ld_afrag(sHid, kt * 32, 192, lane),
                   ld_bfrag(sFrag, F_W2AF + kt, lane), xf2);
#pragma unroll
    for (int kt = 0; kt < 2; ++kt)
      xb2 = wmma16(ld_afrag(sHid, 128 + kt * 32, 192, lane),
                   ld_bfrag(sFrag, F_WGB + kt, lane), xb2);
    // 5) y2 from unclipped xf2/xb2, then clip everything into the carry
    v8f y2 = phaseB(xf2, xb2);
#pragma unroll
    for (int v = 0; v < 8; ++v) {
      xf[v] = fminf(fmaxf(xf2[v], minx),  maxx);
      xb[v] = fminf(fmaxf(xb2[v], minxb), maxxb);
      y[v]  = fminf(fmaxf(y2[v],  miny),  maxy);
    }
  }
}

extern "C" void kernel_launch(void* const* d_in, const int* in_sizes, int n_in,
                              void* d_out, int out_size, void* d_ws, size_t ws_size,
                              hipStream_t stream) {
  (void)in_sizes; (void)n_in; (void)out_size; (void)d_ws; (void)ws_size;
  SimArgs A;
  A.x_f_0   = (const float*)d_in[0];
  A.x_b_0   = (const float*)d_in[1];
  A.u       = (const float*)d_in[2];
  A.theta_f = (const float*)d_in[3];
  A.theta_h = (const float*)d_in[4];
  A.min_xb  = (const float*)d_in[5];
  A.max_xb  = (const float*)d_in[6];
  A.min_x   = (const float*)d_in[7];
  A.max_x   = (const float*)d_in[8];
  A.min_y   = (const float*)d_in[9];
  A.max_y   = (const float*)d_in[10];
  A.u_mean  = (const float*)d_in[11];
  A.u_std   = (const float*)d_in[12];
  A.x_mean  = (const float*)d_in[13];
  A.x_std   = (const float*)d_in[14];
  A.y_mean  = (const float*)d_in[15];
  A.y_std   = (const float*)d_in[16];
  A.fW1  = (const float*)d_in[17]; A.fb1  = (const float*)d_in[18];
  A.fW2  = (const float*)d_in[19]; A.fb2  = (const float*)d_in[20];
  A.fbW1 = (const float*)d_in[21]; A.fbb1 = (const float*)d_in[22];
  A.fbW2 = (const float*)d_in[23]; A.fbb2 = (const float*)d_in[24];
  A.gbW1 = (const float*)d_in[25]; A.gbb1 = (const float*)d_in[26];
  A.gbW2 = (const float*)d_in[27]; A.gbb2 = (const float*)d_in[28];
  A.hW1  = (const float*)d_in[29]; A.hb1  = (const float*)d_in[30];
  A.hW2  = (const float*)d_in[31]; A.hb2  = (const float*)d_in[32];
  A.hbW1 = (const float*)d_in[33]; A.hbb1 = (const float*)d_in[34];
  A.hbW2 = (const float*)d_in[35]; A.hbb2 = (const float*)d_in[36];
  A.oXF = (float*)d_out;
  A.oXB = A.oXF + (size_t)kB * kT * 16;
  A.oY  = A.oXB + (size_t)kB * kT * 16;
  // 128 tiles of 16 batch rows, one wave32 each -> spreads across WGPs
  sim_ssm_wmma_kernel<<<kB / 16, 32, 0, stream>>>(A);
}
